// MPNN_51642686767905
// MI455X (gfx1250) — matrix operand order repaired and checked by reading, not compiled, not run on hardware
//
#include <hip/hip_runtime.h>

// ---------------- problem constants (match reference) ----------------
#define NN 50000      // nodes
#define NE 600000     // edges
#define DD 128        // node feature dim
#define ED 32         // edge feature dim
#define NLAYER 3
#define BN_EPS 1e-5f

// ---------------- WMMA vector types (gfx1250, wave32) ----------------
typedef __attribute__((ext_vector_type(16))) __bf16 v16bf;
typedef __attribute__((ext_vector_type(8)))  float  v8f;

// Native converts: backend emits v_cvt_(pk_)bf16_f32.
__device__ __forceinline__ __bf16 f2bf(float f) { return static_cast<__bf16>(f); }
__device__ __forceinline__ unsigned short f2bf_bits(float f) {
  return __builtin_bit_cast(unsigned short, static_cast<__bf16>(f));
}
__device__ __forceinline__ __bf16 bits2bf(unsigned short s) {
  return __builtin_bit_cast(__bf16, s);
}
// A-fragment K base for 16-bit A 16x32 layout (ISA 7.12.2):
// VGPR v (v<4): K = 2v + 8*half ; (v>=4): K = 16 + 2(v-4) + 8*half
__device__ __forceinline__ int a_k0(int v, int hf) {
  return ((v < 4) ? 2 * v : 16 + 2 * (v - 4)) + 8 * hf;
}

// =====================================================================
// Kernel 1: per-edge  e = edge_attr @ W_lin + b ; msg = relu(x[src]+e);
//           atomic scatter-add msg into agg[dst].
// One wave handles 16 edges; block = 4 waves = 64 edges.
// Phases: (1) 8 back-to-back WMMAs -> full 16x128 e tile in regs,
//         (2) per 16-col tile: batch 8 gather loads (1 wait),
//         (3) fused bias+gather+relu+atomic scatter.
// x and agg (25.6 MB each) stay resident in the 192 MB L2.
// =====================================================================
__global__ void __launch_bounds__(128)
gine_edge_kernel(const float* __restrict__ x,
                 const long long* __restrict__ src,
                 const long long* __restrict__ dst,
                 const float* __restrict__ edge_attr,
                 const float* __restrict__ W_lin,   // [ED][DD] layer slice
                 const float* __restrict__ b_lin,   // [DD]
                 float* __restrict__ agg) {
  __shared__ unsigned short sW[ED * DD];   // W_lin as bf16, 8 KB
  __shared__ int sSrc[4][16];
  __shared__ int sDst[4][16];

  const int tid  = threadIdx.x;
  const int wave = tid >> 5;
  const int lane = tid & 31;

  for (int i = tid; i < ED * DD; i += 128) sW[i] = f2bf_bits(W_lin[i]);

  const int eb = (blockIdx.x * 4 + wave) * 16;   // edge base, 600000%64==0
  if (lane < 16) {
    sSrc[wave][lane] = (int)src[eb + lane];
    sDst[wave][lane] = (int)dst[eb + lane];
  }
  __syncthreads();

  const int mrow = lane & 15;   // A-matrix row carried by this lane
  const int hf   = lane >> 4;   // K-half select

  // warm the gather path (x rows live in L2)
  __builtin_prefetch(&x[(unsigned)sSrc[wave][mrow] * DD], 0, 3);

  // ---- Phase 1: A fragment + 8 WMMAs (full 16x128 e tile) ----
  const float* arow = edge_attr + (size_t)(eb + mrow) * ED;
  v16bf a;
#pragma unroll
  for (int v = 0; v < 8; ++v) {
    const int k0 = a_k0(v, hf);
    a[2 * v]     = f2bf(arow[k0]);
    a[2 * v + 1] = f2bf(arow[k0 + 1]);
  }

  const int kb = 0;   // B-matrix K base; rows follow a_k0 order, col = lane&15
  const v8f czero = {};
  v8f acc[8];
#pragma unroll
  for (int t = 0; t < 8; ++t) {           // 8 N-tiles of 16 cols
    v16bf b;
#pragma unroll
    for (int v = 0; v < 8; ++v) {
      const int n0 = t * 16 + (lane & 15);
      const int k0 = kb + a_k0(v, hf);
      b[2 * v]     = bits2bf(sW[k0 * DD + n0]);
      b[2 * v + 1] = bits2bf(sW[(k0 + 1) * DD + n0]);
    }
    acc[t] = __builtin_amdgcn_wmma_f32_16x16x32_bf16(false, a, false, b,
                                                     (short)0, czero,
                                                     false, false);
  }

  // ---- Phase 2/3: gather + scatter, batched per tile ----
  // C layout: lane holds col = t*16+(lane&15); VGPR r -> row m = r + 8*hf
  const unsigned col0 = (unsigned)(lane & 15);
  unsigned soff[8], doff[8];
#pragma unroll
  for (int r = 0; r < 8; ++r) {
    const int mm = r + 8 * hf;
    soff[r] = (unsigned)sSrc[wave][mm] * (unsigned)DD + col0;
    doff[r] = (unsigned)sDst[wave][mm] * (unsigned)DD + col0;
  }

#pragma unroll
  for (int t = 0; t < 8; ++t) {
    const unsigned tc = (unsigned)(t * 16);
    const float bias = b_lin[tc + col0];
    float xv[8];
#pragma unroll
    for (int r = 0; r < 8; ++r)            // 8 loads issued, single wait
      xv[r] = x[soff[r] + tc];
#pragma unroll
    for (int r = 0; r < 8; ++r) {
      float msg = acc[t][r] + bias + xv[r];
      msg = msg > 0.f ? msg : 0.f;
      atomicAdd(agg + (doff[r] + tc), msg);  // global_atomic_add_f32 (L2)
    }
  }
}

// =====================================================================
// Kernel 2: u = relu((x + agg) @ W1 + b1), stored as bf16.
// One wave = 16 rows; block = 5 waves = 80 rows; 625 blocks (exact).
// =====================================================================
__global__ void __launch_bounds__(160)
gine_mlp1_kernel(const float* __restrict__ x,
                 const float* __restrict__ agg,
                 const float* __restrict__ W1,   // [DD][DD]
                 const float* __restrict__ b1,   // [DD]
                 unsigned short* __restrict__ u_bf) {
  __shared__ unsigned short sW[DD * DD];   // 32 KB bf16
  const int tid  = threadIdx.x;
  const int wave = tid >> 5;
  const int lane = tid & 31;

  for (int i = tid; i < DD * DD; i += 160) sW[i] = f2bf_bits(W1[i]);

  const int rb = (blockIdx.x * 5 + wave) * 16;   // 50000 % 80 == 0
  __syncthreads();

  const int mrow = lane & 15;
  const int hf   = lane >> 4;
  const float* xr = x   + (unsigned)(rb + mrow) * DD;
  const float* gr = agg + (unsigned)(rb + mrow) * DD;

  const v8f czero = {};
  v8f acc[8];
#pragma unroll
  for (int t = 0; t < 8; ++t) acc[t] = czero;

#pragma unroll
  for (int kc = 0; kc < 4; ++kc) {         // K = 128 in 4 chunks of 32
    v16bf a;
#pragma unroll
    for (int v = 0; v < 8; ++v) {
      const int k0 = kc * 32 + a_k0(v, hf);
      a[2 * v]     = f2bf(xr[k0]     + gr[k0]);
      a[2 * v + 1] = f2bf(xr[k0 + 1] + gr[k0 + 1]);
    }
    const int kb = kc * 32;
#pragma unroll
    for (int t = 0; t < 8; ++t) {
      v16bf b;
#pragma unroll
      for (int v = 0; v < 8; ++v) {
        const int n0 = t * 16 + (lane & 15);
        const int k0 = kb + a_k0(v, hf);
        b[2 * v]     = bits2bf(sW[k0 * DD + n0]);
        b[2 * v + 1] = bits2bf(sW[(k0 + 1) * DD + n0]);
      }
      acc[t] = __builtin_amdgcn_wmma_f32_16x16x32_bf16(false, a, false, b,
                                                       (short)0, acc[t],
                                                       false, false);
    }
  }
#pragma unroll
  for (int t = 0; t < 8; ++t) {
    const int col  = t * 16 + (lane & 15);
    const float bias = b1[col];
#pragma unroll
    for (int r = 0; r < 8; ++r) {
      const unsigned row = (unsigned)(rb + r + 8 * hf);
      float v = acc[t][r] + bias;
      v = v > 0.f ? v : 0.f;
      u_bf[row * (unsigned)DD + (unsigned)col] = f2bf_bits(v);
    }
  }
}

// =====================================================================
// Kernel 3: z = u @ W2 + b2 (f32 out), plus per-column sum / sum-of-squares
// for BatchNorm, reduced in-wave via shfl_xor(16) then atomics.
// =====================================================================
__global__ void __launch_bounds__(160)
gine_mlp2_kernel(const unsigned short* __restrict__ u_bf,
                 const float* __restrict__ W2,   // [DD][DD]
                 const float* __restrict__ b2,   // [DD]
                 float* __restrict__ z,
                 float* __restrict__ sums,
                 float* __restrict__ sumsq) {
  __shared__ unsigned short sW[DD * DD];
  const int tid  = threadIdx.x;
  const int wave = tid >> 5;
  const int lane = tid & 31;

  for (int i = tid; i < DD * DD; i += 160) sW[i] = f2bf_bits(W2[i]);

  const int rb = (blockIdx.x * 5 + wave) * 16;
  __syncthreads();

  const int mrow = lane & 15;
  const int hf   = lane >> 4;
  const unsigned short* ur = u_bf + (unsigned)(rb + mrow) * DD;

  const v8f czero = {};
  v8f acc[8];
#pragma unroll
  for (int t = 0; t < 8; ++t) acc[t] = czero;

#pragma unroll
  for (int kc = 0; kc < 4; ++kc) {
    v16bf a;
#pragma unroll
    for (int v = 0; v < 8; ++v) {
      const int k0 = kc * 32 + a_k0(v, hf);
      a[2 * v]     = bits2bf(ur[k0]);
      a[2 * v + 1] = bits2bf(ur[k0 + 1]);
    }
    const int kb = kc * 32;
#pragma unroll
    for (int t = 0; t < 8; ++t) {
      v16bf b;
#pragma unroll
      for (int v = 0; v < 8; ++v) {
        const int n0 = t * 16 + (lane & 15);
        const int k0 = kb + a_k0(v, hf);
        b[2 * v]     = bits2bf(sW[k0 * DD + n0]);
        b[2 * v + 1] = bits2bf(sW[(k0 + 1) * DD + n0]);
      }
      acc[t] = __builtin_amdgcn_wmma_f32_16x16x32_bf16(false, a, false, b,
                                                       (short)0, acc[t],
                                                       false, false);
    }
  }
#pragma unroll
  for (int t = 0; t < 8; ++t) {
    const int col  = t * 16 + (lane & 15);
    const float bias = b2[col];
    float s0 = 0.f, s1 = 0.f;
#pragma unroll
    for (int r = 0; r < 8; ++r) {
      const unsigned row = (unsigned)(rb + r + 8 * hf);
      const float zv = acc[t][r] + bias;
      z[row * (unsigned)DD + (unsigned)col] = zv;
      s0 += zv;
      s1 += zv * zv;
    }
    // lanes l and l^16 hold the same column (other 8 rows): combine
    s0 += __shfl_xor(s0, 16);
    s1 += __shfl_xor(s1, 16);
    if (hf == 0) {
      atomicAdd(&sums[col],  s0);
      atomicAdd(&sumsq[col], s1);
    }
  }
}

// =====================================================================
// Kernel 4a: fold batch stats into per-column scale/shift (1 block).
//   scale = gamma * rsqrt(var + eps);  shift = beta - mu * scale
// =====================================================================
__global__ void __launch_bounds__(DD)
gine_bn_prep_kernel(const float* __restrict__ sums,
                    const float* __restrict__ sumsq,
                    const float* __restrict__ gamma,
                    const float* __restrict__ beta,
                    float* __restrict__ scale,
                    float* __restrict__ shift) {
  const int col = threadIdx.x;
  const float invN = 1.0f / (float)NN;
  const float mu  = sums[col] * invN;
  const float var = sumsq[col] * invN - mu * mu;
  const float sc  = gamma[col] * rsqrtf(var + BN_EPS);
  scale[col] = sc;
  shift[col] = beta[col] - mu * sc;
}

// =====================================================================
// Kernel 4b: apply BN + ReLU, float4-vectorized streaming pass.
// =====================================================================
__global__ void __launch_bounds__(256)
gine_bn_apply_kernel(const float* __restrict__ z,
                     const float* __restrict__ scale,
                     const float* __restrict__ shift,
                     float* __restrict__ out) {
  const unsigned i4 = blockIdx.x * 256u + threadIdx.x;   // NN*DD/4 exact
  const unsigned cb = (i4 * 4u) & (DD - 1u);
  const float4 zv = ((const float4*)z)[i4];
  const float4 sc = *(const float4*)(scale + cb);
  const float4 sh = *(const float4*)(shift + cb);
  float4 o;
  o.x = fmaxf(fmaf(zv.x, sc.x, sh.x), 0.f);
  o.y = fmaxf(fmaf(zv.y, sc.y, sh.y), 0.f);
  o.z = fmaxf(fmaf(zv.z, sc.z, sh.z), 0.f);
  o.w = fmaxf(fmaf(zv.w, sc.w, sh.w), 0.f);
  ((float4*)out)[i4] = o;
}

// =====================================================================
// Host orchestration (graph-capture safe: stream ops only)
// =====================================================================
extern "C" void kernel_launch(void* const* d_in, const int* in_sizes, int n_in,
                              void* d_out, int out_size, void* d_ws, size_t ws_size,
                              hipStream_t stream) {
  const float*     x_in  = (const float*)d_in[0];
  const long long* eidx  = (const long long*)d_in[1];   // int64 [2][NE]
  const float*     eattr = (const float*)d_in[2];
  const float*     W_lin = (const float*)d_in[3];       // [3][ED][DD]
  const float*     b_lin = (const float*)d_in[4];       // [3][DD]
  const float*     W1    = (const float*)d_in[5];       // [3][DD][DD]
  const float*     b1    = (const float*)d_in[6];
  const float*     W2    = (const float*)d_in[7];
  const float*     b2    = (const float*)d_in[8];
  const float*     gamma = (const float*)d_in[9];
  const float*     beta  = (const float*)d_in[10];

  const size_t NB = (size_t)NN * DD * sizeof(float);    // 25.6 MB
  char* ws = (char*)d_ws;
  float*          x_cur = (float*)ws;                      // [NN][DD] f32
  float*          agg   = (float*)(ws + NB);               // [NN][DD] f32 (reused as z)
  unsigned short* u_bf  = (unsigned short*)(ws + 2 * NB);  // [NN][DD] bf16
  float*          sums  = (float*)(ws + 2 * NB + (size_t)NN * DD * 2);
  float*          sumsq = sums  + DD;
  float*          scale = sumsq + DD;
  float*          shift = scale + DD;

  hipMemcpyAsync(x_cur, x_in, NB, hipMemcpyDeviceToDevice, stream);

  const long long* src = eidx;
  const long long* dst = eidx + NE;

  for (int l = 0; l < NLAYER; ++l) {
    hipMemsetAsync(agg,  0, NB, stream);
    hipMemsetAsync(sums, 0, 2 * DD * sizeof(float), stream);

    gine_edge_kernel<<<NE / 64, 128, 0, stream>>>(
        x_cur, src, dst, eattr,
        W_lin + (size_t)l * ED * DD, b_lin + (size_t)l * DD, agg);

    gine_mlp1_kernel<<<NN / 80, 160, 0, stream>>>(
        x_cur, agg, W1 + (size_t)l * DD * DD, b1 + (size_t)l * DD, u_bf);

    // agg is dead after mlp1 -> reuse it as z
    gine_mlp2_kernel<<<NN / 80, 160, 0, stream>>>(
        u_bf, W2 + (size_t)l * DD * DD, b2 + (size_t)l * DD,
        agg, sums, sumsq);

    gine_bn_prep_kernel<<<1, DD, 0, stream>>>(sums, sumsq,
        gamma + (size_t)l * DD, beta + (size_t)l * DD, scale, shift);

    float* outp = (l == NLAYER - 1) ? (float*)d_out : x_cur;
    gine_bn_apply_kernel<<<(NN * DD / 4) / 256, 256, 0, stream>>>(
        agg, scale, shift, outp);
  }
}
